// SheetBase_69406671503656
// MI455X (gfx1250) — compile-verified, matrix operand order checked by reading, not run on hardware
//
#include <hip/hip_runtime.h>
#include <hip/hip_bf16.h>
#include <math.h>

typedef _Float16 v16h __attribute__((ext_vector_type(16)));
typedef _Float16 v8h  __attribute__((ext_vector_type(8)));
typedef float    v8f  __attribute__((ext_vector_type(8)));

#define TABLE_LOG2 19
#define TABLE_SIZE (1u << TABLE_LOG2)
#define N_LEVELS   16
#define CLAMP_V    25.133f
#define TPB        256
#define PTS_PER_BLK 64

// ---------------------------------------------------------------------------
// Pack W[Nout, K] (row-major f32) into WMMA-B fragment order (f16).
//   dst[((nt*KT + kt)*32 + lane)*16 + h] = W[o, k]
//   o = 16*nt + (lane & 15)
//   k = 32*kt + (h & 7) + 8*(lane >> 4) + 16*(h >> 3)
// ---------------------------------------------------------------------------
__global__ void pack_w_kernel(const float* __restrict__ W, int K, int Nout,
                              _Float16* __restrict__ dst) {
  int tid = blockIdx.x * TPB + threadIdx.x;
  if (tid >= K * Nout) return;
  int h    = tid & 15;
  int lane = (tid >> 4) & 31;
  int frag = tid >> 9;
  int KT   = K >> 5;
  int kt   = frag % KT;
  int nt   = frag / KT;
  int k = 32 * kt + (h & 7) + 8 * (lane >> 4) + 16 * (h >> 3);
  int o = 16 * nt + (lane & 15);
  dst[tid] = (_Float16)W[(size_t)o * K + k];
}

// ---------------------------------------------------------------------------
// 4 adjacent 16x16 output tiles per wave: each A fragment reused 4x.
// ---------------------------------------------------------------------------
__device__ __forceinline__ void gemm_tile4(const _Float16* __restrict__ Ain,
                                           int rowBase, int kdim,
                                           const _Float16* __restrict__ Bp,
                                           int KT, int ntBase, int lane,
                                           v8f acc[4]) {
  int arow = (rowBase + (lane & 15)) * kdim + 8 * (lane >> 4);
  const v8h* bp[4];
  v8f z = {};
#pragma unroll
  for (int t = 0; t < 4; ++t) {
    bp[t] = (const v8h*)(Bp + (size_t)(ntBase + t) * KT * 512) + lane * 2;
    acc[t] = z;
  }
  for (int kt = 0; kt < KT; ++kt) {
    v16h a;
    ((v8h*)&a)[0] = *(const v8h*)(Ain + arow + 32 * kt);
    ((v8h*)&a)[1] = *(const v8h*)(Ain + arow + 32 * kt + 16);
#pragma unroll
    for (int t = 0; t < 4; ++t) {
      v16h b;
      ((v8h*)&b)[0] = bp[t][0];
      ((v8h*)&b)[1] = bp[t][1];
      __builtin_prefetch((const void*)(bp[t] + 64), 0, 0);  // global_prefetch_b8
      bp[t] += 64;  // 512 halves per fragment
      acc[t] = __builtin_amdgcn_wmma_f32_16x16x32_f16(
          /*neg_a=*/false, a, /*neg_b=*/false, b,
          /*c_mod=*/(short)0, acc[t], /*reuse_a=*/false, /*reuse_b=*/false);
    }
  }
}

__device__ __forceinline__ float act_sin(float v) {
  v = fminf(fmaxf(v, -CLAMP_V), CLAMP_V);
  return __sinf(v);
}

// D layout (16x16 f32): lane 0-15 -> N=lane, M=r; lane 16-31 -> N=lane-16, M=r+8
__device__ __forceinline__ void store_act4(_Float16* __restrict__ H, int ldH,
                                           int rowBase, int ntBase, int lane,
                                           const v8f acc[4],
                                           const float* __restrict__ bias) {
  int n  = lane & 15;
  int mb = (lane >> 4) * 8;
#pragma unroll
  for (int t = 0; t < 4; ++t) {
    int nt = ntBase + t;
    float bv = bias[nt * 16 + n];
#pragma unroll
    for (int r = 0; r < 8; ++r) {
      float v = act_sin(acc[t][r] + bv);
      H[(rowBase + mb + r) * ldH + nt * 16 + n] = (_Float16)v;
    }
  }
}

// ---------------------------------------------------------------------------
// Fused: hashgrid encode -> 3x (GEMM + sin) ; 64 points per block, 8 waves
// ---------------------------------------------------------------------------
__global__ __launch_bounds__(TPB) void
fused_hashgrid_siren(const int* __restrict__ yv, const int* __restrict__ xv,
                     const float* __restrict__ cam, const float* __restrict__ table,
                     const _Float16* __restrict__ P1, const float* __restrict__ b1,
                     const _Float16* __restrict__ P2, const float* __restrict__ b2,
                     const _Float16* __restrict__ P3, const float* __restrict__ b3,
                     float* __restrict__ out) {
  __shared__ _Float16 h0[PTS_PER_BLK * 64];    //  8 KB : codes(32) | cam(32)
  __shared__ _Float16 hA[PTS_PER_BLK * 512];   // 64 KB
  __shared__ _Float16 hB[PTS_PER_BLK * 512];   // 64 KB
  __shared__ float    camStage[32];            // async-copied camera features
  const int tid  = threadIdx.x;
  const int pt0  = blockIdx.x * PTS_PER_BLK;
  const int wave = tid >> 5;
  const int lane = tid & 31;

  // ---- async copy cam_feat (32 f32) into LDS: ASYNCcnt path ----
  if (wave == 0) {
    unsigned    ldsoff = (unsigned)(uintptr_t)(&camStage[lane]);
    const float* gp    = cam + lane;
    asm volatile("global_load_async_to_lds_b32 %0, %1, off"
                 :: "v"(ldsoff), "v"(gp) : "memory");
    asm volatile("s_wait_asynccnt 0x0" ::: "memory");
  }
  __syncthreads();

  // ---- camera feature broadcast into columns 32..63 ----
  for (int i = tid; i < PTS_PER_BLK * 32; i += TPB) {
    int row = i >> 5, c = i & 31;
    h0[row * 64 + 32 + c] = (_Float16)camStage[c];
  }

  // ---- multires hashgrid encode: 64 pts x 16 levels ----
  for (int task = tid; task < PTS_PER_BLK * N_LEVELS; task += TPB) {
    int pt  = task >> 4;
    int lvl = task & 15;
    float resf = floorf(32.0f * __powf(1.3819f, (float)lvl));
    int   resi = (int)resf;
    float ux = fminf(fmaxf((float)xv[pt0 + pt], 0.0f), 511.0f) * (1.0f / 511.0f);
    float uy = fminf(fmaxf((float)yv[pt0 + pt], 0.0f), 511.0f) * (1.0f / 511.0f);
    float px = ux * resf, py = uy * resf;
    int x0 = min(max((int)floorf(px), 0), resi - 1);
    int y0 = min(max((int)floorf(py), 0), resi - 1);
    float fx = px - (float)x0, fy = py - (float)y0;
    unsigned stride = (unsigned)(resi + 1);
    bool dense = ((long long)stride * (long long)stride) <= (long long)TABLE_SIZE;
    const float* tb = table + (size_t)lvl * TABLE_SIZE * 2;
    float w[4] = {(1.f - fx) * (1.f - fy), fx * (1.f - fy),
                  (1.f - fx) * fy,          fx * fy};
    float c0 = 0.f, c1 = 0.f;
#pragma unroll
    for (int c = 0; c < 4; ++c) {
      unsigned cx = (unsigned)(x0 + (c & 1));
      unsigned cy = (unsigned)(y0 + (c >> 1));
      unsigned idx = dense ? (cx + cy * stride)
                           : ((cx ^ (cy * 2654435761u)) & (TABLE_SIZE - 1u));
      c0 += w[c] * tb[(size_t)idx * 2 + 0];
      c1 += w[c] * tb[(size_t)idx * 2 + 1];
    }
    h0[pt * 64 + 2 * lvl + 0] = (_Float16)c0;
    h0[pt * 64 + 2 * lvl + 1] = (_Float16)c1;
  }
  __syncthreads();

  const int rowBase = (wave >> 1) * 16;  // 4 row-groups of 16 points
  const int colHalf = wave & 1;          // split output tiles between wave pairs

  // ---- layer 1: [64,64] x [64,512] ----
  for (int g = 0; g < 4; ++g) {
    int ntBase = colHalf * 16 + g * 4;
    v8f acc[4];
    gemm_tile4(h0, rowBase, 64, P1, 2, ntBase, lane, acc);
    store_act4(hA, 512, rowBase, ntBase, lane, acc, b1);
  }
  __syncthreads();

  // ---- layer 2: [64,512] x [512,512] ----
  for (int g = 0; g < 4; ++g) {
    int ntBase = colHalf * 16 + g * 4;
    v8f acc[4];
    gemm_tile4(hA, rowBase, 512, P2, 16, ntBase, lane, acc);
    store_act4(hB, 512, rowBase, ntBase, lane, acc, b2);
  }
  __syncthreads();

  // ---- layer 3: [64,512] x [512,384] -> global f32 ----
  for (int g = 0; g < 3; ++g) {
    int ntBase = colHalf * 12 + g * 4;
    v8f acc[4];
    gemm_tile4(hB, rowBase, 512, P3, 16, ntBase, lane, acc);
    int n  = lane & 15;
    int mb = (lane >> 4) * 8;
#pragma unroll
    for (int t = 0; t < 4; ++t) {
      int nt = ntBase + t;
      float bv = b3[nt * 16 + n];
#pragma unroll
      for (int r = 0; r < 8; ++r) {
        float v = act_sin(acc[t][r] + bv);
        out[(size_t)(pt0 + rowBase + mb + r) * 384 + nt * 16 + n] = v;
      }
    }
  }
}

// ---------------------------------------------------------------------------
extern "C" void kernel_launch(void* const* d_in, const int* in_sizes, int n_in,
                              void* d_out, int out_size, void* d_ws, size_t ws_size,
                              hipStream_t stream) {
  const int*   y    = (const int*)d_in[0];
  const int*   x    = (const int*)d_in[1];
  const float* cam  = (const float*)d_in[2];
  const float* tabl = (const float*)d_in[3];
  const float* W1   = (const float*)d_in[4];
  const float* b1   = (const float*)d_in[5];
  const float* W2   = (const float*)d_in[6];
  const float* b2   = (const float*)d_in[7];
  const float* W3   = (const float*)d_in[8];
  const float* b3   = (const float*)d_in[9];
  float*       outp = (float*)d_out;

  _Float16* p1 = (_Float16*)d_ws;          // 64*512   halves
  _Float16* p2 = p1 + 64 * 512;            // 512*512  halves
  _Float16* p3 = p2 + 512 * 512;           // 512*384  halves

  pack_w_kernel<<<(64 * 512) / TPB, TPB, 0, stream>>>(W1, 64, 512, p1);
  pack_w_kernel<<<(512 * 512) / TPB, TPB, 0, stream>>>(W2, 512, 512, p2);
  pack_w_kernel<<<(512 * 384) / TPB, TPB, 0, stream>>>(W3, 512, 384, p3);

  int n_pts  = in_sizes[0];
  int blocks = n_pts / PTS_PER_BLK;  // N = 2^19 -> 8192 blocks
  fused_hashgrid_siren<<<blocks, TPB, 0, stream>>>(
      y, x, cam, tabl, p1, b1, p2, b2, p3, b3, outp);
}
